// UCM_Context_Model_88158498718319
// MI455X (gfx1250) — matrix operand order repaired and checked by reading, not compiled, not run on hardware
//
#include <hip/hip_runtime.h>
#include <hip/hip_bf16.h>

#define CC 32
#define KK 27
#define NPTS 60000
#define MPTS 300000

typedef __attribute__((ext_vector_type(16))) __bf16 v16bf;
typedef __attribute__((ext_vector_type(8)))  __bf16 v8bf;
typedef __attribute__((ext_vector_type(8)))  float  v8f;

// ---------------------------------------------------------------------------
// Pack a row-major 32x32 f32 weight matrix into the WMMA B-operand lane layout
// for v_wmma_f32_16x16x32_bf16 (B is 32x16, two halves per matrix).
//   lane l, element e (0..15):  K = (e&7) + 8*(l/16) + 16*(e/8),  N = l%16
// Packed matrix = 2 halves * 32 lanes * 16 bf16 = 2 KB.
// ---------------------------------------------------------------------------
__global__ void pack_w(const float* __restrict__ W, __bf16* __restrict__ dst) {
    int mat  = blockIdx.x;
    int lane = threadIdx.x;              // 32 threads
    const float* src = W + (size_t)mat * CC * CC;
#pragma unroll
    for (int h = 0; h < 2; ++h) {
        __bf16* d = dst + (((size_t)mat * 2 + h) * 32 + lane) * 16;
#pragma unroll
        for (int e = 0; e < 16; ++e) {
            int Kr = (e & 7) + 8 * (lane >> 4) + 16 * (e >> 3);
            int Nc = h * 16 + (lane & 15);
            d[e] = (__bf16)src[Kr * CC + Nc];
        }
    }
}

// One-time neighbor-table transpose so per-k index loads are contiguous:
//   outT[k*N + i] = in[i*K + k]   (coalesced writes)
__global__ void transpose_idx(const int* __restrict__ in, int* __restrict__ outT,
                              int N, int K) {
    size_t t = (size_t)blockIdx.x * 256 + threadIdx.x;
    if (t >= (size_t)N * K) return;
    size_t k = t / (size_t)N;
    size_t i = t - k * (size_t)N;
    outT[t] = in[i * (size_t)K + k];
}

// ---------------------------------------------------------------------------
// Gather-sconv, bf16 inputs / f32 accumulate:
//   out[i] = act( bias + sum_k fb[nbrT[k,i]] @ W[k] + addsrc[i] )
// fb is pre-converted (and pre-masked) bf16. One wave32 owns 32 rows
// (two 16-row M-tiles): both B halves are reused -> 4 WMMAs per k-slot.
// All packed B matrices for this layer are staged in LDS once per block
// (cooperative copy, dynamic smem = kslots*2048 B), so the inner loop's
// vmem pipe carries only the random gathers while B comes from the ds pipe.
// A lane layout (16-bit A 16x32): lane l%16 = row; lanes 0-15 carry
// K {0..7,16..23}, lanes 16-31 carry K {8..15,24..31} -> two 16B loads.
// Epilogue writes f32 and an optional bf16 shadow scaled by an optional
// per-row write mask (gather source for the next masked sconv).
// ---------------------------------------------------------------------------
__global__ __launch_bounds__(256) void sconv_wmma(
    float* __restrict__ out,
    __bf16* __restrict__ outbf,          // nullable bf16 shadow
    const float* __restrict__ wmaskv,    // nullable row mask for the shadow
    const __bf16* __restrict__ fb,       // gather source (bf16, pre-masked)
    const int* __restrict__ nbrT, int idx_pitch, int kslots,
    const __bf16* __restrict__ Wp,
    const float* __restrict__ bias,
    const float* __restrict__ addsrc,    // nullable residual
    int relu, int N)
{
    extern __shared__ char smem[];
    __bf16* Bs = (__bf16*)smem;

    // ---- stage all packed B matrices for this layer into LDS (all threads,
    //      before any wave-level early exit: barrier safety) ----
    {
        const uint4* gsrc = (const uint4*)Wp;
        uint4*       ldst = (uint4*)Bs;
        int nchunk = kslots * 128;        // 16B chunks
        for (int i = threadIdx.x; i < nchunk; i += 256) ldst[i] = gsrc[i];
    }
    __syncthreads();

    int wave = threadIdx.x >> 5;
    int lane = threadIdx.x & 31;
    int tile = blockIdx.x * 8 + wave;
    int row0 = tile * 32;
    if (row0 < N) {
        int lrow = lane & 15;
        int khi  = lane >> 4;
        int rA   = row0 + lrow;           // rows of M-tile 0
        int rB   = row0 + 16 + lrow;      // rows of M-tile 1

        int n0 = lrow;
        int n1 = 16 + lrow;
        float b0 = bias[n0];
        float b1v = bias[n1];
        v8f acc00, acc01, acc10, acc11;
#pragma unroll
        for (int j = 0; j < 8; ++j) {
            acc00[j] = b0; acc01[j] = b1v;
            acc10[j] = b0; acc11[j] = b1v;
        }

        const int colA = 8 * khi;         // bf16 cols [colA,+8) and [16+colA,+8)
        for (int k = 0; k < kslots; ++k) {
            int idx0 = nbrT[(size_t)k * idx_pitch + rA];
            int idx1 = nbrT[(size_t)k * idx_pitch + rB];
            const __bf16* s0 = fb + (size_t)idx0 * CC;
            const __bf16* s1 = fb + (size_t)idx1 * CC;
            v8bf a0lo = *(const v8bf*)(s0 + colA);
            v8bf a0hi = *(const v8bf*)(s0 + 16 + colA);
            v8bf a1lo = *(const v8bf*)(s1 + colA);
            v8bf a1hi = *(const v8bf*)(s1 + 16 + colA);
            v16bf A0, A1;
#pragma unroll
            for (int j = 0; j < 8; ++j) {
                A0[j] = a0lo[j]; A0[8 + j] = a0hi[j];
                A1[j] = a1lo[j]; A1[8 + j] = a1hi[j];
            }
            const v16bf B0 = *(const v16bf*)(Bs + ((k * 2 + 0) * 32 + lane) * 16);
            const v16bf B1 = *(const v16bf*)(Bs + ((k * 2 + 1) * 32 + lane) * 16);
            acc00 = __builtin_amdgcn_wmma_f32_16x16x32_bf16(false, A0, false, B0,
                                                            (short)0, acc00, false, false);
            acc01 = __builtin_amdgcn_wmma_f32_16x16x32_bf16(false, A0, false, B1,
                                                            (short)0, acc01, false, false);
            acc10 = __builtin_amdgcn_wmma_f32_16x16x32_bf16(false, A1, false, B0,
                                                            (short)0, acc10, false, false);
            acc11 = __builtin_amdgcn_wmma_f32_16x16x32_bf16(false, A1, false, B1,
                                                            (short)0, acc11, false, false);
        }

        // D layout: lane holds column n, VGPR j holds row (j + 8*khi) in tile
#pragma unroll
        for (int t = 0; t < 2; ++t) {
#pragma unroll
            for (int j = 0; j < 8; ++j) {
                int mrow = row0 + t * 16 + 8 * khi + j;
                float v0 = t ? acc10[j] : acc00[j];
                float v1 = t ? acc11[j] : acc01[j];
                if (addsrc) {
                    v0 += addsrc[(size_t)mrow * CC + n0];
                    v1 += addsrc[(size_t)mrow * CC + n1];
                }
                if (relu) {
                    v0 = v0 > 0.f ? v0 : 0.f;
                    v1 = v1 > 0.f ? v1 : 0.f;
                }
                out[(size_t)mrow * CC + n0] = v0;
                out[(size_t)mrow * CC + n1] = v1;
                if (outbf) {
                    float wm = wmaskv ? wmaskv[mrow] : 1.0f;
                    outbf[(size_t)mrow * CC + n0] = (__bf16)(v0 * wm);
                    outbf[(size_t)mrow * CC + n1] = (__bf16)(v1 * wm);
                }
            }
        }
    }
}

// ---------------------------------------------------------------------------
// Head: h -> relu(W1) -> relu(W2) -> logits(16) -> softmax -> bits per point
// ---------------------------------------------------------------------------
__global__ __launch_bounds__(256) void head_bits(
    const float* __restrict__ h,
    const float* __restrict__ W1, const float* __restrict__ b1,
    const float* __restrict__ W2, const float* __restrict__ b2,
    const float* __restrict__ W3, const float* __restrict__ b3,
    const int* __restrict__ gt, const float* __restrict__ mask,
    float* __restrict__ bits, int N)
{
    int i = blockIdx.x * 256 + threadIdx.x;
    if (i >= N) return;
    float x[32], y[32];
#pragma unroll
    for (int c = 0; c < 32; ++c) x[c] = h[(size_t)i * 32 + c];
#pragma unroll
    for (int o = 0; o < 32; ++o) {
        float s = b1[o];
#pragma unroll
        for (int c = 0; c < 32; ++c) s += x[c] * W1[c * 32 + o];
        y[o] = s > 0.f ? s : 0.f;
    }
#pragma unroll
    for (int o = 0; o < 32; ++o) {
        float s = b2[o];
#pragma unroll
        for (int c = 0; c < 32; ++c) s += y[c] * W2[c * 32 + o];
        x[o] = s > 0.f ? s : 0.f;
    }
    float lg[16];
    float mx = -3.4e38f;
#pragma unroll
    for (int o = 0; o < 16; ++o) {
        float s = b3[o];
#pragma unroll
        for (int c = 0; c < 32; ++c) s += x[c] * W3[c * 16 + o];
        lg[o] = s;
        mx = s > mx ? s : mx;
    }
    float sum = 0.f;
#pragma unroll
    for (int o = 0; o < 16; ++o) sum += __expf(lg[o] - mx);
    int g = gt[i];
    float lgg = lg[0];
#pragma unroll
    for (int o = 1; o < 16; ++o) if (o == g) lgg = lg[o];
    float p  = __expf(lgg - mx) / sum;
    float bt = -log2f(p + 1e-10f);
    bt = fminf(fmaxf(bt, 0.f), 50.f);
    bits[i] = bt * mask[i];
}

// Deterministic single-block reduction: acc += sum(v)
__global__ void reduce_add(const float* __restrict__ v, int N, float* __restrict__ acc) {
    __shared__ float s[256];
    float t = 0.f;
    for (int i = threadIdx.x; i < N; i += 256) t += v[i];
    s[threadIdx.x] = t;
    __syncthreads();
    for (int st = 128; st > 0; st >>= 1) {
        if ((int)threadIdx.x < st) s[threadIdx.x] += s[threadIdx.x + st];
        __syncthreads();
    }
    if (threadIdx.x == 0) *acc += s[0];
}

// ---------------------------------------------------------------------------
// Elementwise producers (each emits the bf16 gather-shadow where needed)
// ---------------------------------------------------------------------------
__global__ void gather_emb_bf(const float* __restrict__ emb, const int* __restrict__ idx,
                              __bf16* __restrict__ outbf, int N) {
    int t = blockIdx.x * 256 + threadIdx.x;
    if (t >= N * 32) return;
    outbf[t] = (__bf16)emb[(size_t)idx[t >> 5] * 32 + (t & 31)];
}

__global__ void prep_mask(const int* __restrict__ coords, const int* __restrict__ occ,
                          float* __restrict__ m1, float* __restrict__ m2,
                          int* __restrict__ g0, int* __restrict__ g1, int N) {
    int i = blockIdx.x * 256 + threadIdx.x;
    if (i >= N) return;
    int ps = coords[i * 4 + 1] + coords[i * 4 + 2] + coords[i * 4 + 3];
    float a = ((ps & 1) == 0) ? 1.f : 0.f;
    m1[i] = a;
    m2[i] = 1.f - a;
    int o = occ[i];
    g0[i] = o & 15;
    g1[i] = o >> 4;
}

// out = f + emb[gt];  shadow = (out * wmask) as bf16
__global__ void add_emb(const float* __restrict__ f, const float* __restrict__ emb,
                        const int* __restrict__ gt, float* __restrict__ out,
                        __bf16* __restrict__ outbf, const float* __restrict__ wmask, int N) {
    int t = blockIdx.x * 256 + threadIdx.x;
    if (t >= N * 32) return;
    int i = t >> 5;
    float v = f[t] + emb[(size_t)gt[i] * 32 + (t & 31)];
    out[t] = v;
    outbf[t] = (__bf16)(v * wmask[i]);
}

// shadow-only: (f + emb[occ]) * mask -> bf16 (only ever used as gather source)
__global__ void masked_feats_bf(const float* __restrict__ f, const float* __restrict__ emb,
                                const int* __restrict__ occ, const float* __restrict__ m,
                                __bf16* __restrict__ outbf, int N) {
    int t = blockIdx.x * 256 + threadIdx.x;
    if (t >= N * 32) return;
    int i = t >> 5;
    outbf[t] = (__bf16)((f[t] + emb[(size_t)occ[i] * 32 + (t & 31)]) * m[i]);
}

__global__ void fuse_k(const float* __restrict__ a, const float* __restrict__ g,
                       const float* __restrict__ W, const float* __restrict__ b,
                       float* __restrict__ out, __bf16* __restrict__ outbf,
                       const float* __restrict__ wmask, int N) {
    int t = blockIdx.x * 256 + threadIdx.x;
    if (t >= N * 32) return;
    int i = t >> 5, o = t & 31;
    float s = b[o];
#pragma unroll
    for (int c = 0; c < 32; ++c) s += a[(size_t)i * 32 + c] * W[c * 32 + o];
#pragma unroll
    for (int c = 0; c < 32; ++c) s += g[(size_t)i * 32 + c] * W[(32 + c) * 32 + o];
    s = s > 0.f ? s : 0.f;
    out[t] = s;
    outbf[t] = (__bf16)(s * wmask[i]);
}

__global__ void init_acc(float* acc) { if (threadIdx.x == 0) *acc = 0.f; }
__global__ void finalize_k(const float* acc, float* out) {
    if (threadIdx.x == 0) out[0] = *acc / (float)MPTS;
}

// ---------------------------------------------------------------------------
extern "C" void kernel_launch(void* const* d_in, const int* in_sizes, int n_in,
                              void* d_out, int out_size, void* d_ws, size_t ws_size,
                              hipStream_t stream) {
    const float* prior_emb  = (const float*)d_in[0];
    const float* tgt_embW   = (const float*)d_in[1];
    const float* tgt_embb   = (const float*)d_in[2];
    const float* prior_convW= (const float*)d_in[3];
    const float* prior_convb= (const float*)d_in[4];
    const float* tgt_convW  = (const float*)d_in[5];
    const float* tgt_convb  = (const float*)d_in[6];
    const float* grp_convW  = (const float*)d_in[7];
    const float* grp_convb  = (const float*)d_in[8];
    const float* nbrW       = (const float*)d_in[9];
    const float* nbrb       = (const float*)d_in[10];
    const float* fuseW      = (const float*)d_in[11];
    const float* fuseb      = (const float*)d_in[12];
    const float* headW1     = (const float*)d_in[13];
    const float* headb1     = (const float*)d_in[14];
    const float* headW2     = (const float*)d_in[15];
    const float* headb2     = (const float*)d_in[16];
    const float* headW3     = (const float*)d_in[17];
    const float* headb3     = (const float*)d_in[18];
    const float* s1_emb     = (const float*)d_in[19];
    const int*   x_O        = (const int*)d_in[20];
    const int*   parent_idx = (const int*)d_in[21];
    const int*   coords     = (const int*)d_in[22];
    const int*   occ        = (const int*)d_in[23];
    const int*   nbr_p      = (const int*)d_in[24];
    const int*   nbr_c      = (const int*)d_in[25];
    float* out = (float*)d_out;

    // ---- workspace carve ----
    char* ws = (char*)d_ws;
    size_t off = 0;
    auto carve = [&](size_t bytes) -> void* {
        void* p = ws + off;
        off = (off + bytes + 255) & ~(size_t)255;
        return p;
    };
    const int NMAT = 514; // tgt_embW(1) + prior(135) + tgt(135) + grp(216) + nbr(27)
    __bf16* packed = (__bf16*)carve((size_t)NMAT * 2048);
    float*  P0 = (float*)carve((size_t)NPTS * CC * 4);
    float*  P1 = (float*)carve((size_t)NPTS * CC * 4);
    float*  P2 = (float*)carve((size_t)NPTS * CC * 4);
    __bf16* PbfA = (__bf16*)carve((size_t)NPTS * CC * 2);
    __bf16* PbfB = (__bf16*)carve((size_t)NPTS * CC * 2);
    float*  MB0 = (float*)carve((size_t)MPTS * CC * 4);
    float*  MB1 = (float*)carve((size_t)MPTS * CC * 4);
    float*  MB2 = (float*)carve((size_t)MPTS * CC * 4);
    float*  MB3 = (float*)carve((size_t)MPTS * CC * 4);
    __bf16* MbfA = (__bf16*)carve((size_t)MPTS * CC * 2);
    __bf16* MbfB = (__bf16*)carve((size_t)MPTS * CC * 2);
    int*   nbrT_p = (int*)carve((size_t)NPTS * KK * 4);
    int*   nbrT_c = (int*)carve((size_t)MPTS * KK * 4);
    float* mask1 = (float*)carve((size_t)MPTS * 4);
    float* mask2 = (float*)carve((size_t)MPTS * 4);
    int*   gt0   = (int*)carve((size_t)MPTS * 4);
    int*   gt1   = (int*)carve((size_t)MPTS * 4);
    float* bitsv = (float*)carve((size_t)MPTS * 4);
    float* acc   = (float*)carve(256);

    // matrix bases in the packed buffer
    const int MB_TGT_EMB = 0, MB_PRIOR = 1, MB_TGT = 136, MB_GRP = 271, MB_NBR = 487;

    // ---- pack weights to WMMA bf16 B layout ----
    pack_w<<<1,   32, 0, stream>>>(tgt_embW,    packed + (size_t)MB_TGT_EMB * 1024);
    pack_w<<<135, 32, 0, stream>>>(prior_convW, packed + (size_t)MB_PRIOR * 1024);
    pack_w<<<135, 32, 0, stream>>>(tgt_convW,   packed + (size_t)MB_TGT * 1024);
    pack_w<<<216, 32, 0, stream>>>(grp_convW,   packed + (size_t)MB_GRP * 1024);
    pack_w<<<27,  32, 0, stream>>>(nbrW,        packed + (size_t)MB_NBR * 1024);

    // ---- one-time neighbor-table transposes (coalesced per-k index loads) ----
    {
        size_t np = (size_t)NPTS * KK, nc = (size_t)MPTS * KK;
        transpose_idx<<<(int)((np + 255) / 256), 256, 0, stream>>>(nbr_p, nbrT_p, NPTS, KK);
        transpose_idx<<<(int)((nc + 255) / 256), 256, 0, stream>>>(nbr_c, nbrT_c, MPTS, KK);
    }

    auto sconv = [&](float* o, __bf16* obf, const float* wmask,
                     const __bf16* fb, const int* nbrT, int pitch, int kslots,
                     int matbase, const float* bias, const float* add,
                     int relu, int N) {
        int tiles  = N / 32;              // N is a multiple of 32 here
        int blocks = (tiles + 7) / 8;
        size_t smem = (size_t)kslots * 2048;
        sconv_wmma<<<blocks, 256, smem, stream>>>(o, obf, wmask, fb, nbrT, pitch, kslots,
                                                  packed + (size_t)matbase * 1024,
                                                  bias, add, relu, N);
    };
    auto headcall = [&](const float* h, int hi, const int* gt, const float* msk) {
        int blocks = (MPTS + 255) / 256;
        head_bits<<<blocks, 256, 0, stream>>>(h,
            headW1 + (size_t)hi * 32 * 32, headb1 + (size_t)hi * 32,
            headW2 + (size_t)hi * 32 * 32, headb2 + (size_t)hi * 32,
            headW3 + (size_t)hi * 32 * 16, headb3 + (size_t)hi * 16,
            gt, msk, bitsv, MPTS);
        reduce_add<<<1, 256, 0, stream>>>(bitsv, MPTS, acc);
    };

    init_acc<<<1, 32, 0, stream>>>(acc);

    const int EBP = (NPTS * 32 + 255) / 256;
    const int EBM = (MPTS * 32 + 255) / 256;
    const int PBM = (MPTS + 255) / 256;

    prep_mask<<<PBM, 256, 0, stream>>>(coords, occ, mask1, mask2, gt0, gt1, MPTS);

    // ---- prior path (NP) ----
    gather_emb_bf<<<EBP, 256, 0, stream>>>(prior_emb, x_O, PbfA, NPTS);
    sconv(P1, PbfB, nullptr, PbfA, nbrT_p, NPTS, KK, MB_PRIOR + 0*KK, prior_convb + 0*32, nullptr, 1, NPTS);
    sconv(P2, PbfA, nullptr, PbfB, nbrT_p, NPTS, KK, MB_PRIOR + 1*KK, prior_convb + 1*32, nullptr, 1, NPTS);
    sconv(P0, PbfB, nullptr, PbfA, nbrT_p, NPTS, KK, MB_PRIOR + 2*KK, prior_convb + 2*32, P1,      1, NPTS);
    sconv(P2, PbfA, nullptr, PbfB, nbrT_p, NPTS, KK, MB_PRIOR + 3*KK, prior_convb + 3*32, nullptr, 1, NPTS);
    sconv(P1, PbfB, nullptr, PbfA, nbrT_p, NPTS, KK, MB_PRIOR + 4*KK, prior_convb + 4*32, P0,      1, NPTS);
    // prior feats (bf16) now in PbfB

    // ---- parent embed (M): relu(prior[parent_idx] @ tgt_embW + b), K=1 sconv ----
    sconv(MB0, MbfA, nullptr, PbfB, parent_idx, MPTS, 1, MB_TGT_EMB, tgt_embb, nullptr, 1, MPTS);

    // ---- tgt conv path (M) ----
    sconv(MB1, MbfB, nullptr, MbfA, nbrT_c, MPTS, KK, MB_TGT + 0*KK, tgt_convb + 0*32, nullptr, 1, MPTS);
    sconv(MB2, MbfA, nullptr, MbfB, nbrT_c, MPTS, KK, MB_TGT + 1*KK, tgt_convb + 1*32, nullptr, 1, MPTS);
    sconv(MB0, MbfB, nullptr, MbfA, nbrT_c, MPTS, KK, MB_TGT + 2*KK, tgt_convb + 2*32, MB1,     1, MPTS);
    sconv(MB2, MbfA, nullptr, MbfB, nbrT_c, MPTS, KK, MB_TGT + 3*KK, tgt_convb + 3*32, nullptr, 1, MPTS);
    // feats (f32) -> MB1 ; bf16 shadow pre-masked by mask1 for grp gathers -> MbfB
    sconv(MB1, MbfB, mask1,   MbfA, nbrT_c, MPTS, KK, MB_TGT + 4*KK, tgt_convb + 4*32, MB0,     1, MPTS);

    // ---- grp block 0 (mask1) + head0(gt_s0) ----
    sconv(MB0, MbfA, mask1, MbfB, nbrT_c, MPTS, KK, MB_GRP + 0*KK, grp_convb + 0*32, nullptr, 1, MPTS);
    sconv(MB2, nullptr, nullptr, MbfA, nbrT_c, MPTS, KK, MB_GRP + 1*KK, grp_convb + 1*32, nullptr, 0, MPTS);
    headcall(MB2, 0, gt0, mask1);

    // ---- f1 = feats + s1_emb[0][gt_s0]; grp block 1 + head1(gt_s1) ----
    add_emb<<<EBM, 256, 0, stream>>>(MB1, s1_emb, gt0, MB0, MbfA, mask1, MPTS);
    sconv(MB2, MbfB, mask1, MbfA, nbrT_c, MPTS, KK, MB_GRP + 2*KK, grp_convb + 2*32, nullptr, 1, MPTS);
    sconv(MB3, nullptr, nullptr, MbfB, nbrT_c, MPTS, KK, MB_GRP + 3*KK, grp_convb + 3*32, nullptr, 0, MPTS);
    headcall(MB3, 1, gt1, mask1);

    // ---- masked agg + fuse ----
    masked_feats_bf<<<EBM, 256, 0, stream>>>(MB1, prior_emb, occ, mask1, MbfA, MPTS);
    sconv(MB2, nullptr, nullptr, MbfA, nbrT_c, MPTS, KK, MB_NBR, nbrb, nullptr, 0, MPTS);
    fuse_k<<<EBM, 256, 0, stream>>>(MB1, MB2, fuseW, fuseb, MB0, MbfB, mask2, MPTS);
    // fused (f32) in MB0 ; bf16 pre-masked by mask2 in MbfB

    // ---- grp block 2 (mask2) + head2(gt_s0) ----
    sconv(MB2, MbfA, mask2, MbfB, nbrT_c, MPTS, KK, MB_GRP + 4*KK, grp_convb + 4*32, nullptr, 1, MPTS);
    sconv(MB3, nullptr, nullptr, MbfA, nbrT_c, MPTS, KK, MB_GRP + 5*KK, grp_convb + 5*32, nullptr, 0, MPTS);
    headcall(MB3, 2, gt0, mask2);

    // ---- f2 = fused + s1_emb[1][gt_s0]; grp block 3 + head3(gt_s1) ----
    add_emb<<<EBM, 256, 0, stream>>>(MB0, s1_emb + 16 * 32, gt0, MB2, MbfB, mask2, MPTS);
    sconv(MB2, MbfA, mask2, MbfB, nbrT_c, MPTS, KK, MB_GRP + 6*KK, grp_convb + 6*32, nullptr, 1, MPTS);
    sconv(MB3, nullptr, nullptr, MbfA, nbrT_c, MPTS, KK, MB_GRP + 7*KK, grp_convb + 7*32, nullptr, 0, MPTS);
    headcall(MB3, 3, gt1, mask2);

    finalize_k<<<1, 32, 0, stream>>>(acc, out);
}
